// Relative_Attention_2267742732292
// MI455X (gfx1250) — matrix operand order repaired and checked by reading, not compiled, not run on hardware
//
#include <hip/hip_runtime.h>
#include <cstddef>

// ---------------------------------------------------------------------------
// Problem constants
// ---------------------------------------------------------------------------
#define IMG_HW   577            // N_TOK
#define DIM      512
#define HEADS    8
#define HDIM     64
#define BATCH    32
#define NPAD     608            // 577 padded to multiple of 32 (19*32)
#define MROWS    (BATCH * IMG_HW)   // 18464 = 577 * 32
#define MTILES32 (MROWS / 32)       // 577

// ---------------------------------------------------------------------------
// WMMA types (gfx1250, wave32)
// ---------------------------------------------------------------------------
typedef __bf16 bf16x16 __attribute__((ext_vector_type(16)));
typedef float  f32x8   __attribute__((ext_vector_type(8)));

union Frag {
    bf16x16 v;
    uint4   q[2];
};

__device__ __forceinline__ unsigned short f2bf(float f) {
    unsigned u = __builtin_bit_cast(unsigned, f);
    unsigned r = u + 0x7FFFu + ((u >> 16) & 1u);   // round-to-nearest-even
    return (unsigned short)(r >> 16);
}

// A-fragment: 16x32 bf16 tile, row-major source, rowptr = &A[row][k0].
// Lane (half, m): elems 0..15 -> K = half*16 + i
__device__ __forceinline__ bf16x16 load_frag_a(const unsigned short* rowptr, int half) {
    Frag f;
    f.q[0] = *(const uint4*)(rowptr + half * 16);
    f.q[1] = *(const uint4*)(rowptr + half * 16 + 8);
    return f.v;
}

// B-fragment: 32x16 bf16 tile of W^T where W is row-major [N,K].
// p = &W[n][k0 + half*16]; 16 contiguous bf16 along K.
__device__ __forceinline__ bf16x16 load_frag_b(const unsigned short* p) {
    Frag f;
    f.q[0] = *(const uint4*)(p);
    f.q[1] = *(const uint4*)(p + 8);
    return f.v;
}

__device__ __forceinline__ f32x8 wmma_bf16(bf16x16 a, bf16x16 b, f32x8 c) {
    return __builtin_amdgcn_wmma_f32_16x16x32_bf16(
        /*neg_a=*/false, a, /*neg_b=*/false, b,
        /*c_mod=*/(short)0, c, /*reuse_a=*/false, /*reuse_b=*/false);
}

// ---------------------------------------------------------------------------
// Kernel 1: fp32 -> bf16 conversion
// ---------------------------------------------------------------------------
__global__ void cvt_bf16_kernel(const float* __restrict__ src,
                                unsigned short* __restrict__ dst, int n) {
    int i = blockIdx.x * blockDim.x + threadIdx.x;
    if (i < n) dst[i] = f2bf(src[i]);
}

// ---------------------------------------------------------------------------
// Kernel 2: zero the padded token rows of q/k and padded key columns of v^T
// ---------------------------------------------------------------------------
__global__ void zero_pad_kernel(unsigned short* __restrict__ qb,
                                unsigned short* __restrict__ kb,
                                unsigned short* __restrict__ vt) {
    int bh = blockIdx.x;                       // 0 .. B*H-1
    const int PAD = NPAD - IMG_HW;             // 31
    for (int idx = threadIdx.x; idx < PAD * HDIM; idx += blockDim.x) {
        int r = IMG_HW + idx / HDIM;
        int c = idx % HDIM;
        size_t o = ((size_t)bh * NPAD + r) * HDIM + c;
        qb[o] = 0; kb[o] = 0;
    }
    for (int idx = threadIdx.x; idx < HDIM * PAD; idx += blockDim.x) {
        int r = idx / PAD;
        int c = IMG_HW + idx % PAD;
        vt[((size_t)bh * HDIM + r) * NPAD + c] = 0;
    }
}

// ---------------------------------------------------------------------------
// Kernel 3: QKV GEMM.  qkv[m, c] = sum_k x[m,k] * qkv_w[c,k]
//   block = 128 threads (4 waves); each wave: 32 rows x 64 cols
//   (2 A-frags + 4 B-frags -> 8 WMMA per k-step)
//   grid = (MTILES32, 1536/256)
// Epilogue scatters into q [B,H,NPAD,64] (x0.125), k [B,H,NPAD,64],
// and v transposed [B,H,64,NPAD].
// ---------------------------------------------------------------------------
__global__ __launch_bounds__(128)
void qkv_gemm_kernel(const unsigned short* __restrict__ xb,
                     const unsigned short* __restrict__ wb,
                     unsigned short* __restrict__ qb,
                     unsigned short* __restrict__ kb,
                     unsigned short* __restrict__ vt) {
    const int wave = threadIdx.x >> 5;
    const int lane = threadIdx.x & 31;
    const int half = lane >> 4;
    const int lx   = lane & 15;
    const int row0 = blockIdx.x * 32;
    const int col0 = blockIdx.y * 256 + wave * 64;

    f32x8 acc[2][4] = {};
    const unsigned short* arow0 = xb + (size_t)(row0 + lx) * DIM;
    const unsigned short* arow1 = xb + (size_t)(row0 + 16 + lx) * DIM;
    for (int k0 = 0; k0 < DIM; k0 += 32) {
        bf16x16 a0 = load_frag_a(arow0 + k0, half);
        bf16x16 a1 = load_frag_a(arow1 + k0, half);
        #pragma unroll
        for (int s = 0; s < 4; ++s) {
            bf16x16 b = load_frag_b(wb + (size_t)(col0 + s * 16 + lx) * DIM
                                       + k0 + half * 16);
            acc[0][s] = wmma_bf16(a0, b, acc[0][s]);
            acc[1][s] = wmma_bf16(a1, b, acc[1][s]);
        }
    }

    #pragma unroll
    for (int s = 0; s < 4; ++s) {
        const int c     = col0 + s * 16 + lx;
        const int which = c >> 9;          // 0=q, 1=k, 2=v
        const int rem   = c & 511;
        const int h     = rem >> 6;
        const int dd    = rem & 63;
        #pragma unroll
        for (int rb = 0; rb < 2; ++rb) {
            #pragma unroll
            for (int r = 0; r < 8; ++r) {
                int m     = row0 + rb * 16 + r + 8 * half;
                int bidx  = m / IMG_HW;
                int ntok  = m - bidx * IMG_HW;
                size_t bh = (size_t)bidx * HEADS + h;
                float v   = acc[rb][s][r];
                if (which == 0)
                    qb[(bh * NPAD + ntok) * HDIM + dd] = f2bf(v * 0.125f);
                else if (which == 1)
                    kb[(bh * NPAD + ntok) * HDIM + dd] = f2bf(v);
                else
                    vt[(bh * HDIM + dd) * NPAD + ntok] = f2bf(v);
            }
        }
    }
}

// ---------------------------------------------------------------------------
// Kernel 4: attention.  One workgroup (4 waves) per (b*h, 16-query tile).
//   grid = (B*H, 37)
// ---------------------------------------------------------------------------
__global__ __launch_bounds__(128)
void attn_kernel(const unsigned short* __restrict__ qb,
                 const unsigned short* __restrict__ kb,
                 const unsigned short* __restrict__ vt,
                 const float* __restrict__ bias_table,
                 const int* __restrict__ rel_index,
                 unsigned short* __restrict__ attn_out) {
    __shared__ float          S[16 * NPAD];     // raw scores (f32)
    __shared__ unsigned short Pb[16 * NPAD];    // exp(scores - max) in bf16
    __shared__ float          red[128];
    __shared__ float          rowsum[16];

    const int bh   = blockIdx.x;
    const int h    = bh & (HEADS - 1);
    const int b    = bh >> 3;
    const int row0 = blockIdx.y * 16;           // query tile base (0..576)
    const int wave = threadIdx.x >> 5;
    const int lane = threadIdx.x & 31;
    const int half = lane >> 4;
    const int lx   = lane & 15;

    const unsigned short* qbase = qb + (size_t)bh * NPAD * HDIM;
    const unsigned short* kbase = kb + (size_t)bh * NPAD * HDIM;
    const unsigned short* vbase = vt + (size_t)bh * HDIM * NPAD;

    // ---- Stage 1: S = (q*scale) @ k^T + bias, masked ----
    // q fragments are invariant across key-column tiles: load once.
    const unsigned short* qrow = qbase + (size_t)(row0 + lx) * HDIM;
    bf16x16 qa0 = load_frag_a(qrow, half);        // K = 0..31
    bf16x16 qa1 = load_frag_a(qrow + 32, half);   // K = 32..63

    for (int ct = wave; ct < NPAD / 16; ct += 4) {
        f32x8 acc = {};
        const unsigned short* krow = kbase + (size_t)(ct * 16 + lx) * HDIM
                                           + half * 16;
        acc = wmma_bf16(qa0, load_frag_b(krow), acc);
        acc = wmma_bf16(qa1, load_frag_b(krow + 32), acc);

        const int col = ct * 16 + lx;
        #pragma unroll
        for (int r = 0; r < 8; ++r) {
            int   lr = r + 8 * half;
            int   m  = row0 + lr;
            float v  = acc[r];
            if (col < IMG_HW) {
                if (m < IMG_HW)
                    v += bias_table[rel_index[m * IMG_HW + col] * HEADS + h];
            } else {
                v = -3.0e38f;                    // mask padded keys
            }
            S[lr * NPAD + col] = v;
        }
    }
    __syncthreads();

    // ---- Stage 2: row max, exp, row sum (normalization deferred) ----
    const int row = threadIdx.x >> 3;            // 0..15
    const int sub = threadIdx.x & 7;             // 0..7
    float mx = -3.0e38f;
    for (int c = sub; c < NPAD; c += 8) mx = fmaxf(mx, S[row * NPAD + c]);
    red[threadIdx.x] = mx;
    __syncthreads();
    float rmax = red[row * 8];
    #pragma unroll
    for (int i = 1; i < 8; ++i) rmax = fmaxf(rmax, red[row * 8 + i]);
    float psum = 0.f;
    for (int c = sub; c < NPAD; c += 8) {
        float e = __expf(S[row * NPAD + c] - rmax);
        Pb[row * NPAD + c] = f2bf(e);
        psum += e;
    }
    __syncthreads();
    red[threadIdx.x] = psum;
    __syncthreads();
    if (sub == 0) {
        float s = 0.f;
        #pragma unroll
        for (int i = 0; i < 8; ++i) s += red[row * 8 + i];
        rowsum[row] = s;
    }
    __syncthreads();

    // ---- Stage 3: out = (P @ V) / rowsum; wave owns 16 of 64 head dims ----
    f32x8 acc = {};
    for (int kt = 0; kt < NPAD / 32; ++kt) {
        const int k0 = kt * 32;
        bf16x16 a  = load_frag_a(&Pb[lx * NPAD + k0], half);
        bf16x16 bf = load_frag_b(vbase + (size_t)(wave * 16 + lx) * NPAD
                                       + k0 + half * 16);
        acc = wmma_bf16(a, bf, acc);
    }
    #pragma unroll
    for (int r = 0; r < 8; ++r) {
        int lr   = r + 8 * half;
        int ntok = row0 + lr;
        if (ntok < IMG_HW) {
            int dd = wave * 16 + lx;
            attn_out[((size_t)b * IMG_HW + ntok) * DIM + h * HDIM + dd] =
                f2bf(acc[r] / rowsum[lr]);
        }
    }
}

// ---------------------------------------------------------------------------
// Kernel 5: output projection.  out[m,c] = sum_k a[m,k]*proj_w[c,k] + b[c]
//   block = 128 threads (4 waves); each wave: 32 rows x 64 cols
//   grid = (MTILES32, 512/256)
// ---------------------------------------------------------------------------
__global__ __launch_bounds__(128)
void proj_gemm_kernel(const unsigned short* __restrict__ ab,
                      const unsigned short* __restrict__ wb,
                      const float* __restrict__ bias,
                      float* __restrict__ out) {
    const int wave = threadIdx.x >> 5;
    const int lane = threadIdx.x & 31;
    const int half = lane >> 4;
    const int lx   = lane & 15;
    const int row0 = blockIdx.x * 32;
    const int col0 = blockIdx.y * 256 + wave * 64;

    f32x8 acc[2][4] = {};
    const unsigned short* arow0 = ab + (size_t)(row0 + lx) * DIM;
    const unsigned short* arow1 = ab + (size_t)(row0 + 16 + lx) * DIM;
    for (int k0 = 0; k0 < DIM; k0 += 32) {
        bf16x16 a0 = load_frag_a(arow0 + k0, half);
        bf16x16 a1 = load_frag_a(arow1 + k0, half);
        #pragma unroll
        for (int s = 0; s < 4; ++s) {
            bf16x16 b = load_frag_b(wb + (size_t)(col0 + s * 16 + lx) * DIM
                                       + k0 + half * 16);
            acc[0][s] = wmma_bf16(a0, b, acc[0][s]);
            acc[1][s] = wmma_bf16(a1, b, acc[1][s]);
        }
    }
    #pragma unroll
    for (int s = 0; s < 4; ++s) {
        const int c  = col0 + s * 16 + lx;
        const float pb = bias[c];
        #pragma unroll
        for (int rb = 0; rb < 2; ++rb) {
            #pragma unroll
            for (int r = 0; r < 8; ++r) {
                int m = row0 + rb * 16 + r + 8 * half;
                out[(size_t)m * DIM + c] = acc[rb][s][r] + pb;
            }
        }
    }
}

// ---------------------------------------------------------------------------
// Host-side launcher
// ---------------------------------------------------------------------------
extern "C" void kernel_launch(void* const* d_in, const int* in_sizes, int n_in,
                              void* d_out, int out_size, void* d_ws, size_t ws_size,
                              hipStream_t stream) {
    (void)in_sizes; (void)n_in; (void)out_size; (void)ws_size;

    const float* x          = (const float*)d_in[0];   // [B, N, C]
    const float* qkv_w      = (const float*)d_in[1];   // [3C, C]
    const float* proj_w     = (const float*)d_in[2];   // [C, C]
    const float* proj_b     = (const float*)d_in[3];   // [C]
    const float* bias_table = (const float*)d_in[4];   // [2209, H]
    const int*   rel_index  = (const int*)d_in[5];     // [N, N]
    float*       out        = (float*)d_out;           // [B, N, C]

    // workspace layout (all 256B aligned)
    constexpr size_t SZ_X    = (size_t)MROWS * DIM * 2;            // 18,907,136
    constexpr size_t SZ_QKVW = (size_t)3 * DIM * DIM * 2;          //  1,572,864
    constexpr size_t SZ_PRJW = (size_t)DIM * DIM * 2;              //    524,288
    constexpr size_t SZ_QKV1 = (size_t)BATCH * HEADS * NPAD * HDIM * 2; // 19,922,944

    char* ws = (char*)d_ws;
    unsigned short* x_bf    = (unsigned short*)(ws);
    unsigned short* qkvw_bf = (unsigned short*)(ws + SZ_X);
    unsigned short* prjw_bf = (unsigned short*)(ws + SZ_X + SZ_QKVW);
    unsigned short* q_bf    = (unsigned short*)(ws + SZ_X + SZ_QKVW + SZ_PRJW);
    unsigned short* k_bf    = (unsigned short*)(ws + SZ_X + SZ_QKVW + SZ_PRJW + SZ_QKV1);
    unsigned short* v_t     = (unsigned short*)(ws + SZ_X + SZ_QKVW + SZ_PRJW + 2 * SZ_QKV1);
    unsigned short* attn_bf = (unsigned short*)(ws + SZ_X + SZ_QKVW + SZ_PRJW + 3 * SZ_QKV1);

    // 1) fp32 -> bf16 conversions
    {
        int n = MROWS * DIM;
        cvt_bf16_kernel<<<(n + 255) / 256, 256, 0, stream>>>(x, x_bf, n);
        n = 3 * DIM * DIM;
        cvt_bf16_kernel<<<(n + 255) / 256, 256, 0, stream>>>(qkv_w, qkvw_bf, n);
        n = DIM * DIM;
        cvt_bf16_kernel<<<(n + 255) / 256, 256, 0, stream>>>(proj_w, prjw_bf, n);
    }

    // 2) zero pad rows/cols of q, k, v^T
    zero_pad_kernel<<<BATCH * HEADS, 256, 0, stream>>>(q_bf, k_bf, v_t);

    // 3) QKV projection (WMMA, 32x64 per wave)
    qkv_gemm_kernel<<<dim3(MTILES32, (3 * DIM) / 256), 128, 0, stream>>>(
        x_bf, qkvw_bf, q_bf, k_bf, v_t);

    // 4) attention with relative bias (WMMA + deferred-normalization softmax)
    attn_kernel<<<dim3(BATCH * HEADS, (IMG_HW + 15) / 16), 128, 0, stream>>>(
        q_bf, k_bf, v_t, bias_table, rel_index, attn_bf);

    // 5) output projection (WMMA, 32x64 per wave) -> fp32
    proj_gemm_kernel<<<dim3(MTILES32, DIM / 256), 128, 0, stream>>>(
        attn_bf, prjw_bf, proj_b, out);
}
